// HeteroSubgraphEncoder_64510408786222
// MI455X (gfx1250) — matrix operand order repaired and checked by reading, not compiled, not run on hardware
//
#include <hip/hip_runtime.h>
#include <hip/hip_bf16.h>

// ---------------------------------------------------------------------------
// HeteroSubgraphEncoder forward for MI455X (gfx1250, wave32, WMMA).
//
// v2: vectorized + bank-conflict-free fragment paths.
//  - Weights staged in LDS *transposed with +8-half row padding* so every
//    WMMA B fragment is two ds_load_b128, lanes spread across all 64 banks.
//  - A tiles padded the same way; A fragments are two ds_load_b128 merged
//    with __builtin_shufflevector.
//  - Edge gathers load float4 (global_load_b128) instead of scalar floats.
// ---------------------------------------------------------------------------

typedef __attribute__((ext_vector_type(16))) _Float16 v16h;
typedef __attribute__((ext_vector_type(8)))  _Float16 v8h;
typedef __attribute__((ext_vector_type(4)))  _Float16 v4h;
typedef __attribute__((ext_vector_type(8)))  float    v8f;
typedef __attribute__((ext_vector_type(4)))  float    vf4;

#define N_NODES   20000
#define N_EDGES   320000
#define ATT_SCALE 0.08838834764831845f  // 1/sqrt(128)

// Padded strides (halves): +8 keeps 16-lane fragment loads on distinct banks.
#define SXP  136   // 128-wide rows/cols
#define SSCP 264   // 256-deep scorer columns
#define SANP 40    // 32-deep anchor columns

__device__ __forceinline__ void atomicMaxFloat(float* addr, float val) {
    if (val >= 0.0f) atomicMax((int*)addr, __float_as_int(val));
    else             atomicMin((unsigned int*)addr, (unsigned int)__float_as_int(val));
}

// Build a 16x32 f16 A fragment from a padded LDS row (ISA 7.12.2 layout):
// lane m=lane&15 holds row m; kb=(lane>>4)*8; element j -> k = kb+(j<8?j:8+j).
__device__ __forceinline__ v16h load_a_frag(const _Float16* rowp, int kb) {
    v8h lo = *(const v8h*)(rowp + kb);        // j = 0..7   (k = kb+0..7)
    v8h hi = *(const v8h*)(rowp + kb + 16);   // j = 8..15  (k = kb+16..23)
    return __builtin_shufflevector(lo, hi, 0, 1, 2, 3, 4, 5, 6, 7,
                                   8, 9, 10, 11, 12, 13, 14, 15);
}
// B fragment (KxN, transposed LDS): halves k0+0..7 and k0+16..23 at col*stride + k0.
__device__ __forceinline__ v16h load_b_frag(const _Float16* colp) {
    v8h lo = *(const v8h*)(colp);
    v8h hi = *(const v8h*)(colp + 16);
    return __builtin_shufflevector(lo, hi, 0, 1, 2, 3, 4, 5, 6, 7,
                                   8, 9, 10, 11, 12, 13, 14, 15);
}

// ---------------------------------------------------------------------------
// Y[N,128] = X[N,128] @ W[128,128] + bias   (ACCUM: Y += ...)
// ---------------------------------------------------------------------------
template <bool ACCUM>
__global__ __launch_bounds__(256)
void gemm_n128(const float* __restrict__ X, const float* __restrict__ W,
               const float* __restrict__ bias, float* __restrict__ Y, int N)
{
    __shared__ __attribute__((aligned(16))) _Float16 sX[128 * SXP]; // row-major, padded
    __shared__ __attribute__((aligned(16))) _Float16 sW[128 * SXP]; // TRANSPOSED: [col][k]
    const int tid      = threadIdx.x;
    const int row_base = blockIdx.x * 128;

    for (int i = tid; i < 128 * 32; i += 256) {     // float4 granularity
        const int r = i >> 5, cq = (i & 31) << 2;   // r = row/k, cq = col*4
        vf4 wv = *(const vf4*)&W[r * 128 + cq];     // W[k][c..c+3]
#pragma unroll
        for (int t = 0; t < 4; ++t) sW[(cq + t) * SXP + r] = (_Float16)wv[t];
        const int gr = row_base + r;
        v4h xh = {};
        if (gr < N) {
            vf4 xv = *(const vf4*)&X[(size_t)gr * 128 + cq];
#pragma unroll
            for (int t = 0; t < 4; ++t) xh[t] = (_Float16)xv[t];
        }
        *(v4h*)&sX[r * SXP + cq] = xh;
    }
    __syncthreads();

    const int wv = tid >> 5, lane = tid & 31;
    const int m = lane & 15, hi = lane >> 4;
    const int kb = hi * 8;
    const _Float16* xrow = &sX[(wv * 16 + m) * SXP];

    v16h a[4];
#pragma unroll
    for (int kk = 0; kk < 4; ++kk) a[kk] = load_a_frag(xrow + kk * 32, kb);

#pragma unroll
    for (int nt = 0; nt < 8; ++nt) {
        const int col = nt * 16 + m;
        const _Float16* wcol = &sW[col * SXP + hi * 8];
        v8f acc = {};
#pragma unroll
        for (int kk = 0; kk < 4; ++kk) {
            v16h b = load_b_frag(wcol + kk * 32);
            acc = __builtin_amdgcn_wmma_f32_16x16x32_f16(
                false, a[kk], false, b, (short)0, acc, false, false);
        }
        const float bv = bias[col];
#pragma unroll
        for (int r = 0; r < 8; ++r) {
            int grow = row_base + wv * 16 + r + 8 * hi;
            if (grow < N) {
                size_t o = (size_t)grow * 128 + col;
                float v = acc[r] + bv;
                if (ACCUM) v += Y[o];
                Y[o] = v;
            }
        }
    }
}

// ---------------------------------------------------------------------------
// Fused edge kernel: wave-per-16-edges.
//   s      = leaky_relu(concat(xp_s[src], xp_d[dst]) @ Wsc + bsc)
//   bcoef  = softmax_K32(s)
//   prompt = bcoef @ anchor
//   e      = prompt @ We + be  -> e_out[E,128]
// ---------------------------------------------------------------------------
__global__ __launch_bounds__(256)
void edge_prompt_e(const float* __restrict__ xps, const float* __restrict__ xpd,
                   const int* __restrict__ src, const int* __restrict__ dst,
                   const float* __restrict__ Wsc, const float* __restrict__ bsc,
                   const float* __restrict__ anchor,
                   const float* __restrict__ We, const float* __restrict__ be,
                   float* __restrict__ e_out, int E)
{
    __shared__ __attribute__((aligned(16))) _Float16 sWsc[32 * SSCP];   // [col][k], k<256
    __shared__ __attribute__((aligned(16))) _Float16 sAnc[128 * SANP];  // [col][k], k<32
    __shared__ __attribute__((aligned(16))) _Float16 sWe[128 * SXP];    // [col][k], k<128
    __shared__ __attribute__((aligned(16))) _Float16 sTmp[8][16 * SXP]; // per-wave scratch

    const int tid = threadIdx.x;
    for (int i = tid; i < 256 * 8; i += 256) {          // Wsc: 256x32
        const int k = i >> 3, cq = (i & 7) << 2;
        vf4 w4 = *(const vf4*)&Wsc[k * 32 + cq];
#pragma unroll
        for (int t = 0; t < 4; ++t) sWsc[(cq + t) * SSCP + k] = (_Float16)w4[t];
    }
    for (int i = tid; i < 32 * 32; i += 256) {          // anchor: 32x128
        const int k = i >> 5, cq = (i & 31) << 2;
        vf4 w4 = *(const vf4*)&anchor[k * 128 + cq];
#pragma unroll
        for (int t = 0; t < 4; ++t) sAnc[(cq + t) * SANP + k] = (_Float16)w4[t];
    }
    for (int i = tid; i < 128 * 32; i += 256) {         // We: 128x128
        const int k = i >> 5, cq = (i & 31) << 2;
        vf4 w4 = *(const vf4*)&We[k * 128 + cq];
#pragma unroll
        for (int t = 0; t < 4; ++t) sWe[(cq + t) * SXP + k] = (_Float16)w4[t];
    }
    __syncthreads();

    const int wv = tid >> 5, lane = tid & 31;
    const int m = lane & 15, hi = lane >> 4, kb = hi * 8;
    const int e0 = blockIdx.x * 128 + wv * 16;
    const int em = min(e0 + m, E - 1);
    const int sm = src[em], dm = dst[em];

    // ---- scores = comb @ Wsc ---- (K = 256; chunk kk<4 from xp_s, kk>=4 from xp_d)
    v8f s[2] = {{}, {}};
#pragma unroll
    for (int kk = 0; kk < 8; ++kk) {
        const float* bp = (kk < 4)
            ? xps + (size_t)sm * 128 + kk * 32 + kb
            : xpd + (size_t)dm * 128 + (kk - 4) * 32 + kb;
        vf4 f0 = *(const vf4*)(bp);
        vf4 f1 = *(const vf4*)(bp + 4);
        vf4 f2 = *(const vf4*)(bp + 16);
        vf4 f3 = *(const vf4*)(bp + 20);
        v16h a;
#pragma unroll
        for (int t = 0; t < 4; ++t) {
            a[t]      = (_Float16)f0[t];
            a[4 + t]  = (_Float16)f1[t];
            a[8 + t]  = (_Float16)f2[t];
            a[12 + t] = (_Float16)f3[t];
        }
#pragma unroll
        for (int nt = 0; nt < 2; ++nt) {
            v16h b = load_b_frag(&sWsc[(nt * 16 + m) * SSCP + kk * 32 + hi * 8]);
            s[nt] = __builtin_amdgcn_wmma_f32_16x16x32_f16(
                false, a, false, b, (short)0, s[nt], false, false);
        }
    }

    // ---- bias + leaky_relu + row softmax over 32 cols ----
    const float b0 = bsc[m], b1 = bsc[16 + m];
#pragma unroll
    for (int r = 0; r < 8; ++r) {
        float v0 = s[0][r] + b0; v0 = v0 > 0.f ? v0 : 0.01f * v0;
        float v1 = s[1][r] + b1; v1 = v1 > 0.f ? v1 : 0.01f * v1;
        float mx = fmaxf(v0, v1);
        for (int d = 1; d < 16; d <<= 1) mx = fmaxf(mx, __shfl_xor(mx, d, 16));
        float e0f = __expf(v0 - mx), e1f = __expf(v1 - mx);
        float sum = e0f + e1f;
        for (int d = 1; d < 16; d <<= 1) sum += __shfl_xor(sum, d, 16);
        s[0][r] = e0f / sum;
        s[1][r] = e1f / sum;
    }
    __syncthreads();

    _Float16* sB = &sTmp[wv][0];
#pragma unroll
    for (int r = 0; r < 8; ++r) {               // bcoef rows, padded stride SANP
        sB[(r + 8 * hi) * SANP + m]      = (_Float16)s[0][r];
        sB[(r + 8 * hi) * SANP + 16 + m] = (_Float16)s[1][r];
    }
    __syncthreads();

    // ---- prompt = bcoef @ anchor (K = 32) ----
    v16h ab = load_a_frag(&sB[m * SANP], kb);
    v8f pr[8];
#pragma unroll
    for (int nt = 0; nt < 8; ++nt) {
        v16h b = load_b_frag(&sAnc[(nt * 16 + m) * SANP + hi * 8]);
        v8f z = {};
        pr[nt] = __builtin_amdgcn_wmma_f32_16x16x32_f16(
            false, ab, false, b, (short)0, z, false, false);
    }
    __syncthreads();
#pragma unroll
    for (int nt = 0; nt < 8; ++nt)
#pragma unroll
        for (int r = 0; r < 8; ++r)
            sB[(r + 8 * hi) * SXP + nt * 16 + m] = (_Float16)pr[nt][r];
    __syncthreads();

    // ---- e = prompt @ We + be (K = 128) ----
    v16h ap[4];
#pragma unroll
    for (int kk = 0; kk < 4; ++kk) ap[kk] = load_a_frag(&sB[m * SXP + kk * 32], kb);
#pragma unroll
    for (int nt = 0; nt < 8; ++nt) {
        const _Float16* wcol = &sWe[(nt * 16 + m) * SXP + hi * 8];
        v8f acc = {};
#pragma unroll
        for (int kk = 0; kk < 4; ++kk) {
            v16h b = load_b_frag(wcol + kk * 32);
            acc = __builtin_amdgcn_wmma_f32_16x16x32_f16(
                false, ap[kk], false, b, (short)0, acc, false, false);
        }
        const float bv = be[nt * 16 + m];
#pragma unroll
        for (int r = 0; r < 8; ++r) {
            int er = e0 + r + 8 * hi;
            if (er < E)
                e_out[(size_t)er * 128 + nt * 16 + m] = acc[r] + bv;
        }
    }
}

// ---------------------------------------------------------------------------
// logit[e] = SCALE * dot(q[dst], k[src] + e_row); segment max over dst.
// One wave per edge; each lane reads one float4 per array (b128, coalesced).
// ---------------------------------------------------------------------------
__global__ __launch_bounds__(256)
void edge_logit(const float* __restrict__ qd, const float* __restrict__ ks,
                const float* __restrict__ ebuf, const int* __restrict__ src,
                const int* __restrict__ dst, float* __restrict__ logit,
                float* __restrict__ mx, int E)
{
    const int wv = threadIdx.x >> 5, lane = threadIdx.x & 31;
    const int e = blockIdx.x * 8 + wv;
    if (e >= E) return;
    const int sm = src[e], dm = dst[e];
    vf4 kv = *(const vf4*)&ks[(size_t)sm * 128 + lane * 4];
    vf4 ev = *(const vf4*)&ebuf[(size_t)e * 128 + lane * 4];
    vf4 qv = *(const vf4*)&qd[(size_t)dm * 128 + lane * 4];
    float acc = 0.f;
#pragma unroll
    for (int i = 0; i < 4; ++i) acc += qv[i] * (kv[i] + ev[i]);
    for (int d = 16; d; d >>= 1) acc += __shfl_xor(acc, d, 32);
    if (lane == 0) {
        float lg = acc * ATT_SCALE;
        logit[e] = lg;
        atomicMaxFloat(&mx[dm], lg);
    }
}

__global__ __launch_bounds__(256)
void edge_expsum(float* __restrict__ logit, const int* __restrict__ dst,
                 const float* __restrict__ mx, float* __restrict__ den, int E)
{
    const int e = blockIdx.x * blockDim.x + threadIdx.x;
    if (e >= E) return;
    const int dm = dst[e];
    float w = __expf(logit[e] - mx[dm]);
    logit[e] = w;                 // reuse buffer for exp weights
    atomicAdd(&den[dm], w);
}

__global__ __launch_bounds__(256)
void edge_aggregate(const float* __restrict__ vs, const float* __restrict__ ebuf,
                    const float* __restrict__ wbuf, const float* __restrict__ den,
                    const int* __restrict__ src, const int* __restrict__ dst,
                    float* __restrict__ outd, int E)
{
    const int wv = threadIdx.x >> 5, lane = threadIdx.x & 31;
    const int e = blockIdx.x * 8 + wv;
    if (e >= E) return;
    const int sm = src[e], dm = dst[e];
    const float alpha = wbuf[e] / den[dm];
    vf4 vv = *(const vf4*)&vs[(size_t)sm * 128 + lane * 4];
    vf4 ev = *(const vf4*)&ebuf[(size_t)e * 128 + lane * 4];
#pragma unroll
    for (int i = 0; i < 4; ++i)
        atomicAdd(&outd[(size_t)dm * 128 + lane * 4 + i], alpha * (vv[i] + ev[i]));
}

__global__ void fill_kernel(float* __restrict__ p, float v, size_t n)
{
    size_t i = (size_t)blockIdx.x * blockDim.x + threadIdx.x;
    if (i < n) p[i] = v;
}

__global__ void pool_kernel(const float* __restrict__ x, const int* __restrict__ batch,
                            float* __restrict__ pool, float* __restrict__ cnt, int N)
{
    size_t idx = (size_t)blockIdx.x * blockDim.x + threadIdx.x;
    if (idx >= (size_t)N * 128) return;
    int node = (int)(idx >> 7), c = (int)(idx & 127);
    int g = batch[node];
    atomicAdd(&pool[g * 128 + c], x[idx]);
    if (c == 0) atomicAdd(&cnt[g], 1.0f);
}

// Tiny head: normalize pooled means, average types, relu-MLP, normalize.
__global__ __launch_bounds__(128)
void head_kernel(const float* __restrict__ pa, const float* __restrict__ ca,
                 const float* __restrict__ pb, const float* __restrict__ cb,
                 const float* __restrict__ w0, const float* __restrict__ b0,
                 const float* __restrict__ w1, const float* __restrict__ b1,
                 float* __restrict__ out)
{
    __shared__ float g[8][128];
    __shared__ float h[8][128];
    __shared__ float nrm[16];
    const int t = threadIdx.x;
    if (t < 16) {
        const float* p = t < 8 ? pa : pb;
        const float* c = t < 8 ? ca : cb;
        int gi = t & 7;
        float cnt = fmaxf(c[gi], 1.0f);
        float s = 0.f;
        for (int k = 0; k < 128; ++k) { float v = p[gi * 128 + k] / cnt; s += v * v; }
        nrm[t] = fmaxf(sqrtf(s), 1e-12f);
    }
    __syncthreads();
    for (int gi = 0; gi < 8; ++gi) {
        float va = (pa[gi * 128 + t] / fmaxf(ca[gi], 1.0f)) / nrm[gi];
        float vb = (pb[gi * 128 + t] / fmaxf(cb[gi], 1.0f)) / nrm[8 + gi];
        g[gi][t] = 0.5f * (va + vb);
    }
    __syncthreads();
    for (int gi = 0; gi < 8; ++gi) {
        float acc = b0[t];
        for (int k = 0; k < 128; ++k) acc += g[gi][k] * w0[k * 128 + t];
        h[gi][t] = fmaxf(acc, 0.f);
    }
    __syncthreads();
    for (int gi = 0; gi < 8; ++gi) {
        float acc = b1[t];
        for (int k = 0; k < 128; ++k) acc += h[gi][k] * w1[k * 128 + t];
        g[gi][t] = acc;
    }
    __syncthreads();
    if (t < 8) {
        float s = 0.f;
        for (int k = 0; k < 128; ++k) s += g[t][k] * g[t][k];
        nrm[t] = fmaxf(sqrtf(s), 1e-12f);
    }
    __syncthreads();
    for (int gi = 0; gi < 8; ++gi) out[gi * 128 + t] = g[gi][t] / nrm[gi];
}

// ---------------------------------------------------------------------------
// Host orchestration
// ---------------------------------------------------------------------------
extern "C" void kernel_launch(void* const* d_in, const int* in_sizes, int n_in,
                              void* d_out, int out_size, void* d_ws, size_t ws_size,
                              hipStream_t stream)
{
    (void)in_sizes; (void)n_in; (void)out_size; (void)ws_size;
    const int N = N_NODES, E = N_EDGES;
#define F(i) ((const float*)d_in[(i)])
#define I(i) ((const int*)d_in[(i)])

    const float* x_a = F(0);
    const float* x_b = F(1);
    const int* ei_ab = I(2);
    const int* ei_ba = I(3);
    const int* batch_a = I(4);
    const int* batch_b = I(5);

    // params leaves in jax pytree order (dict keys sorted): head, layers, prompt
    int p = 6;
    const float* hb0 = F(p++); const float* hw0 = F(p++);
    const float* hb1 = F(p++); const float* hw1 = F(p++);
    struct Lin { const float *b, *w; };
    struct EtL { Lin e, k, q, skip, v; };     // keys sorted: e,k,q,skip,v
    EtL L[3][2];
    for (int l = 0; l < 3; ++l)
        for (int et = 0; et < 2; ++et) {      // a__to__b, b__to__a
            EtL& x = L[l][et];
            x.e.b = F(p++);    x.e.w = F(p++);
            x.k.b = F(p++);    x.k.w = F(p++);
            x.q.b = F(p++);    x.q.w = F(p++);
            x.skip.b = F(p++); x.skip.w = F(p++);
            x.v.b = F(p++);    x.v.w = F(p++);
        }
    const float *anch[2], *scb[2], *scw[2];
    for (int et = 0; et < 2; ++et) {          // prompt.et: anchor, scorer{b,w}
        anch[et] = F(p++); scb[et] = F(p++); scw[et] = F(p++);
    }
    Lin proj[2];
    for (int t = 0; t < 2; ++t) { proj[t].b = F(p++); proj[t].w = F(p++); }

    // workspace carve (floats)
    float* ws = (float*)d_ws;
    size_t o = 0;
    auto take = [&](size_t n) { float* r = ws + o; o += n; return r; };
    float* xc[2] = { take((size_t)N * 128), take((size_t)N * 128) };  // cur a,b
    float* xn[2] = { take((size_t)N * 128), take((size_t)N * 128) };  // next a,b
    float* xp[2] = { take((size_t)N * 128), take((size_t)N * 128) };
    float* qb   = take((size_t)N * 128);
    float* kbuf = take((size_t)N * 128);
    float* vbuf = take((size_t)N * 128);
    float* ebuf = take((size_t)E * 128);
    float* lbuf = take((size_t)E);
    float* mxb  = take((size_t)N);
    float* dnb  = take((size_t)N);
    float* pool[2] = { take(8 * 128), take(8 * 128) };
    float* cnt[2]  = { take(8), take(8) };

    hipMemcpyAsync(xc[0], x_a, (size_t)N * 128 * sizeof(float),
                   hipMemcpyDeviceToDevice, stream);
    hipMemcpyAsync(xc[1], x_b, (size_t)N * 128 * sizeof(float),
                   hipMemcpyDeviceToDevice, stream);

    const int gN   = (N + 127) / 128;               // 157
    const int gE   = (E + 127) / 128;               // 2500
    const int gEw  = (E + 7) / 8;                   // wave-per-edge kernels
    const int gFil = (int)(((size_t)N * 128 + 255) / 256);
    const int* src_et[2] = { ei_ab, ei_ba };

    for (int l = 0; l < 3; ++l) {
        gemm_n128<false><<<gN, 256, 0, stream>>>(xc[0], proj[0].w, proj[0].b, xp[0], N);
        gemm_n128<false><<<gN, 256, 0, stream>>>(xc[1], proj[1].w, proj[1].b, xp[1], N);
        fill_kernel<<<gFil, 256, 0, stream>>>(xn[0], 0.f, (size_t)N * 128);
        fill_kernel<<<gFil, 256, 0, stream>>>(xn[1], 0.f, (size_t)N * 128);

        for (int et = 0; et < 2; ++et) {
            const int s = (et == 0) ? 0 : 1, d = 1 - s;
            const int* src = src_et[et];
            const int* dst = src_et[et] + E;
            const EtL& P = L[l][et];

            gemm_n128<false><<<gN, 256, 0, stream>>>(xc[d], P.q.w, P.q.b, qb, N);
            gemm_n128<false><<<gN, 256, 0, stream>>>(xc[s], P.k.w, P.k.b, kbuf, N);
            gemm_n128<false><<<gN, 256, 0, stream>>>(xc[s], P.v.w, P.v.b, vbuf, N);

            edge_prompt_e<<<gE, 256, 0, stream>>>(xp[s], xp[d], src, dst,
                                                  scw[et], scb[et], anch[et],
                                                  P.e.w, P.e.b, ebuf, E);

            fill_kernel<<<(N + 255) / 256, 256, 0, stream>>>(mxb, -3.402823466e38f, (size_t)N);
            fill_kernel<<<(N + 255) / 256, 256, 0, stream>>>(dnb, 0.f, (size_t)N);

            edge_logit<<<gEw, 256, 0, stream>>>(qb, kbuf, ebuf, src, dst, lbuf, mxb, E);
            edge_expsum<<<(E + 255) / 256, 256, 0, stream>>>(lbuf, dst, mxb, dnb, E);
            edge_aggregate<<<gEw, 256, 0, stream>>>(vbuf, ebuf, lbuf, dnb, src, dst, xn[d], E);

            gemm_n128<true><<<gN, 256, 0, stream>>>(xc[d], P.skip.w, P.skip.b, xn[d], N);
        }
        float* t0 = xc[0]; float* t1 = xc[1];
        xc[0] = xn[0]; xc[1] = xn[1];
        xn[0] = t0;    xn[1] = t1;
    }

    fill_kernel<<<(8 * 128 + 255) / 256, 256, 0, stream>>>(pool[0], 0.f, 8 * 128);
    fill_kernel<<<(8 * 128 + 255) / 256, 256, 0, stream>>>(pool[1], 0.f, 8 * 128);
    fill_kernel<<<1, 256, 0, stream>>>(cnt[0], 0.f, 8);
    fill_kernel<<<1, 256, 0, stream>>>(cnt[1], 0.f, 8);
    pool_kernel<<<gFil, 256, 0, stream>>>(xc[0], batch_a, pool[0], cnt[0], N);
    pool_kernel<<<gFil, 256, 0, stream>>>(xc[1], batch_b, pool[1], cnt[1], N);

    head_kernel<<<1, 128, 0, stream>>>(pool[0], cnt[0], pool[1], cnt[1],
                                       hw0, hb0, hw1, hb1, (float*)d_out);
#undef F
#undef I
}
